// ConsciousnessIntegrationNetwork_44753559225119
// MI455X (gfx1250) — matrix operand order repaired and checked by reading, not compiled, not run on hardware
//
#include <hip/hip_runtime.h>

typedef __attribute__((ext_vector_type(16))) _Float16 v16h;
typedef __attribute__((ext_vector_type(8)))  float    v8f;
typedef __attribute__((ext_vector_type(4)))  unsigned v4u;

#if defined(__has_builtin)
#  if __has_builtin(__builtin_amdgcn_tensor_load_to_lds) && __has_builtin(__builtin_amdgcn_s_wait_tensorcnt)
#    define HAVE_TDM 1
#  endif
#endif
#ifndef HAVE_TDM
#  define HAVE_TDM 0
#endif

#define DD 128
#define HH 128
#define WW 128
#define NPOS (2 * DD * HH * WW)   // 4194304 spatial points
#define C1 8
#define C2 16

// ---- WMMA f16 fragment K-maps (wave32, 16x16x32) ----
__device__ __forceinline__ int kmapA(int v, int half) {   // A 16x32 (ISA 7.12.2)
  return ((v & 4) << 2) + ((v & 3) << 1) + (half << 3);
}
__device__ __forceinline__ int kmapB(int v, int half) {   // B 32x16 (same K order as A)
  return ((v & 4) << 2) + ((v & 3) << 1) + (half << 3);
}
__device__ __forceinline__ unsigned pack2(_Float16 lo, _Float16 hi) {
  union { _Float16 h[2]; unsigned u; } t; t.h[0] = lo; t.h[1] = hi; return t.u;
}
__device__ __forceinline__ int lindex(int b, int z, int y, int x) {
  return (((((b << 7) + z) << 7) + y) << 7) + x;
}

// Block tile: 8 y-stacked waves x 32 x-positions at one (b,z) -> 256 outputs.
// LDS slab: [dz:3][yrel:10][xi:34][C] f16, halo zero-filled, + zero-pad tail for K padding.
__device__ __forceinline__ int offA1(int k, int pad) {        // conv1: k = tap, C=1
  if (k >= 27) return pad;
  int dz = k / 9, dy = (k / 3) % 3, dx = k % 3;
  return ((dz * 10 + dy) * 34 + dx) * 2;
}
__device__ __forceinline__ int offA2(int k, int pad) {        // conv2: k = tap*8 + c
  if (k >= 216) return pad;
  int tap = k >> 3, c = k & 7;
  int dz = tap / 9, dy = (tap / 3) % 3, dx = tap % 3;
  return (((dz * 10 + dy) * 34 + dx) * 8 + c) * 2;
}
__device__ __forceinline__ int offA3(int k, int pad) {        // conv3: k = tap*16 + c
  int tap = k >> 4, c = k & 15;
  if (tap >= 27) return pad;
  int dz = tap / 9, dy = (tap / 3) % 3, dx = tap % 3;
  return (((dz * 10 + dy) * 34 + dx) * 16 + c) * 2;
}

// Decode block -> (b, z, y-group, x-tile of 32)
#define BLK_DECODE()                                   \
  const int tid  = threadIdx.x;                        \
  const int lane = tid & 31;                           \
  const int wv   = tid >> 5;                           \
  const int half = lane >> 4;                          \
  const int n    = lane & 15;                          \
  const int xt   = blockIdx.x & 3;                     \
  const int yg   = (blockIdx.x >> 2) & 15;             \
  const int z    = (blockIdx.x >> 6) & 127;            \
  const int b    = blockIdx.x >> 13;                   \
  const int y    = (yg << 3) + wv;                     \
  const int x0   = xt << 5;

#if HAVE_TDM
typedef __attribute__((ext_vector_type(4))) unsigned tdm_v4u;
typedef __attribute__((ext_vector_type(8))) int      tdm_v8i;
typedef __attribute__((ext_vector_type(4))) int      tdm_v4i;
// 3D tile load via Tensor Data Mover (D# per ISA 8.3/8.4).  Upper-edge halo is
// zero-filled by clamped tensor dims (OOB reads return zero).
__device__ __forceinline__ void tdm_load3d(unsigned lds_addr, unsigned long long gaddr,
                                           unsigned t0, unsigned t1, unsigned t2,
                                           unsigned tile0, unsigned tile1, unsigned tile2,
                                           unsigned long long s0, unsigned long long s1) {
  tdm_v4u g0;
  g0[0] = 1u;                                            // count=1, user descriptor
  g0[1] = lds_addr;                                      // LDS byte address
  g0[2] = (unsigned)(gaddr & 0xFFFFFFFFu);               // global_addr[31:0]
  g0[3] = (unsigned)((gaddr >> 32) & 0x01FFFFFFu) | (2u << 30);  // addr[56:32] | type=2
  tdm_v8i g1;
  g1[0] = (int)(1u << 16);                               // data_size = 2B, wg_mask=0
  g1[1] = (int)((t0 & 0xFFFFu) << 16);                   // tensor_dim0[15:0]
  g1[2] = (int)(((t0 >> 16) & 0xFFFFu) | ((t1 & 0xFFFFu) << 16));
  g1[3] = (int)(((t1 >> 16) & 0xFFFFu) | (tile0 << 16)); // tile_dim0
  g1[4] = (int)(tile1 | (tile2 << 16));                  // tile_dim1 | tile_dim2
  g1[5] = (int)(s0 & 0xFFFFFFFFu);                       // tensor_dim0_stride[31:0]
  g1[6] = (int)(((s0 >> 32) & 0xFFFFu) | ((s1 & 0xFFFFu) << 16));
  g1[7] = (int)((s1 >> 16) & 0xFFFFFFFFu);               // tensor_dim1_stride[47:16]
  tdm_v4i g2 = { (int)t2, 0, 0, 0 };                     // tensor_dim2, no dim3
  tdm_v4i g3 = { 0, 0, 0, 0 };
#if __clang_major__ >= 23
  tdm_v8i g4 = { 0, 0, 0, 0, 0, 0, 0, 0 };
  __builtin_amdgcn_tensor_load_to_lds(g0, g1, g2, g3, g4, 0);
#else
  __builtin_amdgcn_tensor_load_to_lds(g0, g1, g2, g3, 0);
#endif
}
#endif

// ---------------- kernel 0: field modulation + f32 -> f16 ----------------
__global__ void k_field(const float* __restrict__ in, _Float16* __restrict__ a0) {
  int i = blockIdx.x * 256 + threadIdx.x;
  int xx = i & 127, yy = (i >> 7) & 127, zz = (i >> 14) & 127;
  float ph = (float)(xx + yy + zz) * (6.28318530718f / 384.0f);
  a0[i] = (_Float16)(in[i] * (1.0f + 0.1f * __sinf(ph)));
}

// ---------------- kernel 1: conv 1->8, K=27 (pad 32), 2 WMMA / wave ----------------
__global__ void k_conv1(const _Float16* __restrict__ a0, const float* __restrict__ w1,
                        const float* __restrict__ b1, _Float16* __restrict__ a1) {
  __shared__ __align__(16) unsigned char tile[2080];   // 3*10*34*2 = 2040, pad @2048
  BLK_DECODE();

#if HAVE_TDM
  if (z >= 1 && yg >= 1 && xt >= 1) {
    if (wv == 0) {
      tdm_load3d((unsigned)(unsigned long long)tile,
                 (unsigned long long)a0 + (unsigned long long)lindex(b, z - 1, (yg << 3) - 1, x0 - 1) * 2,
                 129 - x0, 129 - (yg << 3), 129 - z, 34, 10, 3, 128, 128 * 128);
      __builtin_amdgcn_s_wait_tensorcnt(0);
    }
  } else
#endif
  {
#pragma unroll
    for (int it = 0; it < 4; ++it) {
      int idx = it * 256 + tid;
      if (idx < 1020) {
        _Float16 val = 0;
        int row = idx / 34, xi = idx % 34;
        int dz = row / 10, yr = row % 10;
        int zz = z + dz - 1, yy = (yg << 3) + yr - 1, xx = x0 + xi - 1;
        if ((unsigned)zz < DD && (unsigned)yy < HH && (unsigned)xx < WW)
          val = a0[lindex(b, zz, yy, xx)];
        *(_Float16*)(tile + idx * 2) = val;
      }
    }
  }
  if (tid < 2) *(v4u*)(tile + 2048 + tid * 16) = (v4u){0, 0, 0, 0};

  // B fragment: w1[co=8][tap=27] zero-padded to 32x16 (per-wave registers)
  union { v16h v; unsigned u[8]; } bf;
#pragma unroll
  for (int v = 0; v < 8; ++v) {
    int kb = kmapB(v, half);
    _Float16 lo = 0, hi = 0;
    if (n < C1) {
      if (kb     < 27) lo = (_Float16)w1[n * 27 + kb];
      if (kb + 1 < 27) hi = (_Float16)w1[n * 27 + kb + 1];
    }
    bf.u[v] = pack2(lo, hi);
  }
  __syncthreads();

  const int padA = 2048 - wv * 68;
  const unsigned char* myA = tile + wv * 68 + n * 2;
  union { v16h v; unsigned u[8]; } afA, afB;
#pragma unroll
  for (int v = 0; v < 8; ++v) {
    int k0 = kmapA(v, 0);
    int o0 = half ? offA1(k0 + 8, padA) : offA1(k0, padA);
    int o1 = half ? offA1(k0 + 9, padA) : offA1(k0 + 1, padA);
    int p0 = half ? offA1(k0 + 8, padA - 32) : offA1(k0, padA - 32);
    int p1 = half ? offA1(k0 + 9, padA - 32) : offA1(k0 + 1, padA - 32);
    afA.u[v] = (*(const unsigned short*)(myA + o0)) | ((unsigned)(*(const unsigned short*)(myA + o1)) << 16);
    afB.u[v] = (*(const unsigned short*)(myA + 32 + p0)) | ((unsigned)(*(const unsigned short*)(myA + 32 + p1)) << 16);
  }

  v8f acc0 = {}, acc1 = {};
  acc0 = __builtin_amdgcn_wmma_f32_16x16x32_f16(false, afA.v, false, bf.v, (short)0, acc0, false, false);
  acc1 = __builtin_amdgcn_wmma_f32_16x16x32_f16(false, afB.v, false, bf.v, (short)0, acc1, false, false);

  float bias = (n < C1) ? b1[n] : 0.f;
#pragma unroll
  for (int r = 0; r < 8; ++r) {
    int xr = x0 + r + (half << 3);
    if (n < C1) {
      a1[(size_t)lindex(b, z, y, xr) * C1 + n]      = (_Float16)fmaxf(acc0[r] + bias, 0.f);
      a1[(size_t)lindex(b, z, y, xr + 16) * C1 + n] = (_Float16)fmaxf(acc1[r] + bias, 0.f);
    }
  }
}

// ---------------- kernel 2: conv 8->16, K=216, 14 WMMA / wave ----------------
__global__ void k_conv2(const _Float16* __restrict__ a1, const float* __restrict__ w2,
                        const float* __restrict__ b2, _Float16* __restrict__ a2) {
  __shared__ __align__(32) unsigned ldsB[7][32][8];       // 7 KiB B fragments
  __shared__ __align__(16) unsigned char tile[16576];     // 3*10*34*8*2 = 16320 + 256B pad
  BLK_DECODE();

  {  // cooperative B build
    const int v = wv;
#pragma unroll
    for (int s = 0; s < 7; ++s) {
      int kb = s * 32 + kmapB(v, half);
      _Float16 lo = 0, hi = 0;
      if (kb < 216)     { int t = kb >> 3,       c = kb & 7;       lo = (_Float16)w2[(n * 8 + c) * 27 + t]; }
      if (kb + 1 < 216) { int t = (kb + 1) >> 3, c = (kb + 1) & 7; hi = (_Float16)w2[(n * 8 + c) * 27 + t]; }
      ldsB[s][lane][v] = pack2(lo, hi);
    }
  }

#if HAVE_TDM
  if (z >= 1 && yg >= 1 && xt >= 1) {
    if (wv == 0) {
      tdm_load3d((unsigned)(unsigned long long)tile,
                 (unsigned long long)a1 + (unsigned long long)lindex(b, z - 1, (yg << 3) - 1, x0 - 1) * (C1 * 2),
                 (129 - x0) * C1, 129 - (yg << 3), 129 - z,
                 34 * C1, 10, 3, 128 * C1, 128 * 128 * C1);
      __builtin_amdgcn_s_wait_tensorcnt(0);
    }
  } else
#endif
  {
#pragma unroll
    for (int it = 0; it < 4; ++it) {
      int idx = it * 256 + tid;
      if (idx < 1020) {
        v4u val = {0, 0, 0, 0};
        int row = idx / 34, xi = idx % 34;
        int dz = row / 10, yr = row % 10;
        int zz = z + dz - 1, yy = (yg << 3) + yr - 1, xx = x0 + xi - 1;
        if ((unsigned)zz < DD && (unsigned)yy < HH && (unsigned)xx < WW)
          val = *(const v4u*)(a1 + (size_t)lindex(b, zz, yy, xx) * C1);
        *(v4u*)(tile + idx * 16) = val;
      }
    }
  }
  if (tid < 16) *(v4u*)(tile + 16320 + tid * 16) = (v4u){0, 0, 0, 0};
  __syncthreads();

  const int padA = 16320 - wv * 544;
  const unsigned char* myA = tile + wv * 544 + n * 16;
  v8f acc0 = {}, acc1 = {};
#pragma unroll
  for (int s = 0; s < 7; ++s) {
    union { v16h v; unsigned u[8]; } afA, afB;
#pragma unroll
    for (int v = 0; v < 8; ++v) {
      int k0 = s * 32 + kmapA(v, 0);
      int oa = half ? offA2(k0 + 8, padA)       : offA2(k0, padA);
      int ob = half ? offA2(k0 + 8, padA - 256) : offA2(k0, padA - 256);
      afA.u[v] = *(const unsigned*)(myA + oa);
      afB.u[v] = *(const unsigned*)(myA + 256 + ob);
    }
    v16h bv = *reinterpret_cast<const v16h*>(&ldsB[s][lane][0]);
    acc0 = __builtin_amdgcn_wmma_f32_16x16x32_f16(false, afA.v, false, bv, (short)0, acc0, false, false);
    acc1 = __builtin_amdgcn_wmma_f32_16x16x32_f16(false, afB.v, false, bv, (short)0, acc1, false, false);
  }

  float bias = b2[n];
#pragma unroll
  for (int r = 0; r < 8; ++r) {
    int xr = x0 + r + (half << 3);
    a2[(size_t)lindex(b, z, y, xr) * C2 + n]      = (_Float16)fmaxf(acc0[r] + bias, 0.f);
    a2[(size_t)lindex(b, z, y, xr + 16) * C2 + n] = (_Float16)fmaxf(acc1[r] + bias, 0.f);
  }
}

// ---------------- kernel 3: conv 16->1, channel-major K (k = tap*16 + c), 28 WMMA / wave ----------------
__global__ void k_conv3(const _Float16* __restrict__ a2, const float* __restrict__ w3,
                        const float* __restrict__ b3, float* __restrict__ out) {
  __shared__ __align__(32) unsigned ldsB[14][32][8];      // 14 KiB
  __shared__ __align__(16) unsigned char tile[33152];     // 3*10*34*16*2 = 32640 + 512B pad
  BLK_DECODE();

  {  // B: column n owns channel n  -> D[m,n] = per-channel partial sums
    const int v = wv;
#pragma unroll
    for (int s = 0; s < 14; ++s) {
      int kb = s * 32 + kmapB(v, half);                   // even
      int t = kb >> 4, c = kb & 15;
      _Float16 lo = 0, hi = 0;
      if (t < 27 && c     == n) lo = (_Float16)w3[n * 27 + t];
      if (t < 27 && c + 1 == n) hi = (_Float16)w3[n * 27 + t];
      ldsB[s][lane][v] = pack2(lo, hi);
    }
  }

#if HAVE_TDM
  if (z >= 1 && yg >= 1 && xt >= 1) {
    if (wv == 0) {
      tdm_load3d((unsigned)(unsigned long long)tile,
                 (unsigned long long)a2 + (unsigned long long)lindex(b, z - 1, (yg << 3) - 1, x0 - 1) * (C2 * 2),
                 (129 - x0) * C2, 129 - (yg << 3), 129 - z,
                 34 * C2, 10, 3, 128 * C2, 128 * 128 * C2);
      __builtin_amdgcn_s_wait_tensorcnt(0);
    }
  } else
#endif
  {
#pragma unroll
    for (int it = 0; it < 8; ++it) {
      int idx = it * 256 + tid;
      if (idx < 2040) {
        v4u val = {0, 0, 0, 0};
        int pos = idx >> 1, h8 = idx & 1;
        int row = pos / 34, xi = pos % 34;
        int dz = row / 10, yr = row % 10;
        int zz = z + dz - 1, yy = (yg << 3) + yr - 1, xx = x0 + xi - 1;
        if ((unsigned)zz < DD && (unsigned)yy < HH && (unsigned)xx < WW)
          val = *(const v4u*)(a2 + (size_t)lindex(b, zz, yy, xx) * C2 + h8 * 8);
        *(v4u*)(tile + idx * 16) = val;
      }
    }
  }
  if (tid < 32) *(v4u*)(tile + 32640 + tid * 16) = (v4u){0, 0, 0, 0};
  __syncthreads();

  const int padA = 32640 - wv * 1088;
  const unsigned char* myA = tile + wv * 1088 + n * 32;
  v8f acc0 = {}, acc1 = {};
#pragma unroll
  for (int s = 0; s < 14; ++s) {
    union { v16h v; unsigned u[8]; } afA, afB;
#pragma unroll
    for (int v = 0; v < 8; ++v) {
      int k0 = s * 32 + kmapA(v, 0);
      int oa = half ? offA3(k0 + 8, padA)       : offA3(k0, padA);
      int ob = half ? offA3(k0 + 8, padA - 512) : offA3(k0, padA - 512);
      afA.u[v] = *(const unsigned*)(myA + oa);
      afB.u[v] = *(const unsigned*)(myA + 512 + ob);
    }
    v16h bv = *reinterpret_cast<const v16h*>(&ldsB[s][lane][0]);
    acc0 = __builtin_amdgcn_wmma_f32_16x16x32_f16(false, afA.v, false, bv, (short)0, acc0, false, false);
    acc1 = __builtin_amdgcn_wmma_f32_16x16x32_f16(false, afB.v, false, bv, (short)0, acc1, false, false);
  }

  float bias = b3[0];
#pragma unroll
  for (int r = 0; r < 8; ++r) {
    float s0 = acc0[r], s1 = acc1[r];
    s0 += __shfl_xor(s0, 1, 32);  s1 += __shfl_xor(s1, 1, 32);
    s0 += __shfl_xor(s0, 2, 32);  s1 += __shfl_xor(s1, 2, 32);
    s0 += __shfl_xor(s0, 4, 32);  s1 += __shfl_xor(s1, 4, 32);
    s0 += __shfl_xor(s0, 8, 32);  s1 += __shfl_xor(s1, 8, 32);
    if (n == 0) {
      int xr = x0 + r + (half << 3);
      out[lindex(b, z, y, xr)]      = tanhf(s0 + bias);
      out[lindex(b, z, y, xr + 16)] = tanhf(s1 + bias);
    }
  }
}

extern "C" void kernel_launch(void* const* d_in, const int* in_sizes, int n_in,
                              void* d_out, int out_size, void* d_ws, size_t ws_size,
                              hipStream_t stream) {
  (void)in_sizes; (void)n_in; (void)out_size; (void)ws_size;
  const float* x  = (const float*)d_in[0];
  const float* w1 = (const float*)d_in[1];
  const float* b1 = (const float*)d_in[2];
  const float* w2 = (const float*)d_in[3];
  const float* b2 = (const float*)d_in[4];
  const float* w3 = (const float*)d_in[5];
  const float* b3 = (const float*)d_in[6];
  float* out = (float*)d_out;

  // workspace: act0 (f16, 8 MiB) | act1 (f16 x8ch, 64 MiB) | act2 (f16 x16ch, 128 MiB)
  char* ws = (char*)d_ws;
  _Float16* a0 = (_Float16*)ws;
  _Float16* a1 = (_Float16*)(ws + (size_t)NPOS * 2);
  _Float16* a2 = (_Float16*)(ws + (size_t)NPOS * 2 + (size_t)NPOS * C1 * 2);

  const int nblk = NPOS / 256;     // 16384 blocks: (b,z,y-group,x-tile32), 8 waves x 32 outputs
  k_field<<<NPOS / 256, 256, 0, stream>>>(x, a0);
  k_conv1<<<nblk, 256, 0, stream>>>(a0, w1, b1, a1);
  k_conv2<<<nblk, 256, 0, stream>>>(a1, w2, b2, a2);
  k_conv3<<<nblk, 256, 0, stream>>>(a2, w3, b3, out);
}
